// ScaledDotProductAttention_25151328485506
// MI455X (gfx1250) — matrix operand order repaired and checked by reading, not compiled, not run on hardware
//
#include <hip/hip_runtime.h>
#include <math.h>

typedef __attribute__((ext_vector_type(16))) _Float16 v16h;
typedef __attribute__((ext_vector_type(8)))  _Float16 v8h;
typedef __attribute__((ext_vector_type(8)))  float    v8f;
typedef __attribute__((ext_vector_type(4)))  float    f32x4;

#define BH      32
#define SEQ     2048
#define DH      64
#define NMB     2      // mask batch
#define QTILES  2      // 16-query tiles per wave
#define QB      (NWAVES * QTILES * 16)   // 128 query rows per block
#define NWAVES  4
#define KC      32     // keys per chunk
#define NCHUNK  (SEQ / KC)
#define KROW    72     // padded row stride (f16) of sK: 144B, conflict-free b128
#define VROW    40     // padded row stride (f16) of sV: 80B, conflict-free b128

// Build a 16x32 f16 A-fragment from LDS: elements 0..7 at p[0..7],
// elements 8..15 at p[16..23]  (two aligned ds_load_b128)
__device__ __forceinline__ v16h lds_afrag(const _Float16* __restrict__ p) {
  const v8h lo = *(const v8h*)p;
  const v8h hi = *(const v8h*)(p + 16);
  v16h r;
#pragma unroll
  for (int i = 0; i < 8; ++i) { r[i] = lo[i]; r[8 + i] = hi[i]; }
  return r;
}

// Cooperative staging: thread t handles key = t/4, dims (t%4)*16 .. +15
__device__ __forceinline__ void stage_global_load(const float* __restrict__ Kbh,
                                                  const float* __restrict__ Vbh,
                                                  int kb, int tid,
                                                  f32x4 nk[4], f32x4 nv[4]) {
  const int key = tid >> 2;
  const int dq  = (tid & 3) * 16;
  const float* kg = Kbh + (size_t)(kb + key) * DH + dq;
  const float* vg = Vbh + (size_t)(kb + key) * DH + dq;
#pragma unroll
  for (int i = 0; i < 4; ++i) {
    nk[i] = *(const f32x4*)(kg + 4 * i);
    nv[i] = *(const f32x4*)(vg + 4 * i);
  }
}

__device__ __forceinline__ void stage_lds_store(_Float16* __restrict__ sKb,
                                                _Float16* __restrict__ sVb,
                                                int tid,
                                                const f32x4 nk[4], const f32x4 nv[4]) {
  const int key = tid >> 2;
  const int dq  = (tid & 3) * 16;
  // K: row-major f16, two b128 stores
  _Float16* kd = sKb + key * KROW + dq;
#pragma unroll
  for (int q = 0; q < 2; ++q) {
    v8h h;
#pragma unroll
    for (int i = 0; i < 8; ++i) {
      const f32x4 src = nk[2 * q + (i >> 2)];
      h[i] = (_Float16)src[i & 3];
    }
    *(v8h*)(kd + q * 8) = h;
  }
  // V: transposed into sV[d][key] (scattered b16 stores)
#pragma unroll
  for (int i = 0; i < 16; ++i)
    sVb[(size_t)(dq + i) * VROW + key] = (_Float16)nv[i >> 2][i & 3];
}

__global__ __launch_bounds__(NWAVES * 32)
void fa_fwd_wmma_25151328485506(const float* __restrict__ Qg,
                                const float* __restrict__ Kg,
                                const float* __restrict__ Vg,
                                const float* __restrict__ Mg,
                                float* __restrict__ Og) {
  __shared__ __align__(16) _Float16 sK[2][32 * KROW];
  __shared__ __align__(16) _Float16 sV[2][64 * VROW];

  const int tid  = threadIdx.x;
  const int wave = tid >> 5;
  const int lane = tid & 31;
  const int l16  = lane & 15;
  const int sub  = lane >> 4;   // 16-lane half of the wave

  const int bh    = blockIdx.x / (SEQ / QB);
  const int qblk  = blockIdx.x % (SEQ / QB);
  const int qbase = qblk * QB + wave * (QTILES * 16);

  const float* __restrict__ Kbh = Kg + (size_t)bh * SEQ * DH;
  const float* __restrict__ Vbh = Vg + (size_t)bh * SEQ * DH;

  // Q^T B-fragments, one pair per query tile:
  // lane = query column (qtile base + l16); d = 8*sub..+7, 16+8*sub..+7 (+32 for hi)
  v16h bqlo[QTILES], bqhi[QTILES];
  const float* mrow[QTILES];
#pragma unroll
  for (int t = 0; t < QTILES; ++t) {
    const int q = qbase + 16 * t + l16;
    const float* qrow = Qg + ((size_t)bh * SEQ + q) * DH;
    mrow[t] = Mg + ((size_t)(bh % NMB) * SEQ + q) * SEQ;
    const float* p0 = qrow + sub * 8;
    const float* p1 = qrow + 32 + sub * 8;
#pragma unroll
    for (int i = 0; i < 16; i += 4) {
      const f32x4 a = *(const f32x4*)(p0 + i + (i & 8));
      const f32x4 b = *(const f32x4*)(p1 + i + (i & 8));
#pragma unroll
      for (int j = 0; j < 4; ++j) {
        bqlo[t][i + j] = (_Float16)a[j];
        bqhi[t][i + j] = (_Float16)b[j];
      }
    }
  }

  const v8f zero8 = {0.f, 0.f, 0.f, 0.f, 0.f, 0.f, 0.f, 0.f};
  v8f acc[QTILES][4];                 // O^T accumulators, 4 dim-tiles per qtile
#pragma unroll
  for (int t = 0; t < QTILES; ++t)
#pragma unroll
    for (int dt = 0; dt < 4; ++dt) acc[t][dt] = zero8;
  float m_i[QTILES] = {-__builtin_inff(), -__builtin_inff()};
  float l_i[QTILES] = {0.f, 0.f};

  // Prologue: stage chunk 0 into buffer 0
  f32x4 nk[4], nv[4];
  stage_global_load(Kbh, Vbh, 0, tid, nk, nv);
  stage_lds_store(&sK[0][0], &sV[0][0], tid, nk, nv);
  __syncthreads();

  for (int c = 0; c < NCHUNK; ++c) {
    const int kb  = c * KC;
    const int buf = c & 1;
    const bool has_next = (c + 1) < NCHUNK;

    // Pipelined global loads for the NEXT chunk (latency hidden under compute)
    if (has_next) stage_global_load(Kbh, Vbh, kb + KC, tid, nk, nv);

    // ---- S^T = K_tile x Q^T: shared K fragments feed both query tiles ----
    const _Float16* kr0 = &sK[buf][(size_t)l16 * KROW];
    const _Float16* kr1 = &sK[buf][(size_t)(16 + l16) * KROW];
    const int dstart = 8 * sub;

    const v16h ka00 = lds_afrag(kr0 + dstart);
    const v16h ka01 = lds_afrag(kr0 + 32 + dstart);
    const v16h ka10 = lds_afrag(kr1 + dstart);
    const v16h ka11 = lds_afrag(kr1 + 32 + dstart);

    v8f sc[QTILES][2];
#pragma unroll
    for (int t = 0; t < QTILES; ++t) {
      sc[t][0] = __builtin_amdgcn_wmma_f32_16x16x32_f16(false, ka00, false, bqlo[t], (short)0, zero8,    false, false);
      sc[t][0] = __builtin_amdgcn_wmma_f32_16x16x32_f16(false, ka01, false, bqhi[t], (short)0, sc[t][0], false, false);
      sc[t][1] = __builtin_amdgcn_wmma_f32_16x16x32_f16(false, ka10, false, bqlo[t], (short)0, zero8,    false, false);
      sc[t][1] = __builtin_amdgcn_wmma_f32_16x16x32_f16(false, ka11, false, bqhi[t], (short)0, sc[t][1], false, false);
    }

    // ---- per-tile: mask + online softmax + P^T fragment ----
    v16h pb[QTILES];
#pragma unroll
    for (int t = 0; t < QTILES; ++t) {
      const f32x4 mk0a = *(const f32x4*)(mrow[t] + kb + 8 * sub);
      const f32x4 mk0b = *(const f32x4*)(mrow[t] + kb + 8 * sub + 4);
      const f32x4 mk1a = *(const f32x4*)(mrow[t] + kb + 16 + 8 * sub);
      const f32x4 mk1b = *(const f32x4*)(mrow[t] + kb + 16 + 8 * sub + 4);

      float s0[8], s1[8];
#pragma unroll
      for (int j = 0; j < 4; ++j) {
        s0[j]     = sc[t][0][j]     * 0.125f + mk0a[j];
        s0[j + 4] = sc[t][0][j + 4] * 0.125f + mk0b[j];
        s1[j]     = sc[t][1][j]     * 0.125f + mk1a[j];
        s1[j + 4] = sc[t][1][j + 4] * 0.125f + mk1b[j];
      }

      float tmax = s0[0];
#pragma unroll
      for (int j = 1; j < 8; ++j) tmax = fmaxf(tmax, s0[j]);
#pragma unroll
      for (int j = 0; j < 8; ++j) tmax = fmaxf(tmax, s1[j]);
      tmax = fmaxf(tmax, __shfl_xor(tmax, 16, 32));

      const float m_new = fmaxf(m_i[t], tmax);
      const float corr  = __expf(m_i[t] - m_new);

      float p0[8], p1[8], rs = 0.f;
#pragma unroll
      for (int j = 0; j < 8; ++j) {
        p0[j] = __expf(s0[j] - m_new);
        p1[j] = __expf(s1[j] - m_new);
        rs += p0[j] + p1[j];
      }
      rs += __shfl_xor(rs, 16, 32);
      l_i[t] = l_i[t] * corr + rs;
      m_i[t] = m_new;

#pragma unroll
      for (int dt = 0; dt < 4; ++dt)
#pragma unroll
        for (int j = 0; j < 8; ++j)
          acc[t][dt][j] *= corr;

      // half h needs keys 8h..8h+7 and 16+8h..16+8h+7 of its query (own p0/p1)
#pragma unroll
      for (int j = 0; j < 8; ++j) {
        const float mine = sub ? p1[j] : p0[j];
        const float send = sub ? p0[j] : p1[j];
        const float recv = __shfl_xor(send, 16, 32);
        const float lo   = p0[j];  // K element 8*sub+j
        const float hi   = p1[j];  // K element 16+8*sub+j
        pb[t][j]     = (_Float16)lo;
        pb[t][8 + j] = (_Float16)hi;
      }
    }

    // ---- O^T += V^T x P^T: each V^T fragment feeds both query tiles ----
#pragma unroll
    for (int dt = 0; dt < 4; ++dt) {
      const _Float16* vr = &sV[buf][(size_t)(dt * 16 + l16) * VROW + 8 * sub];
      const v16h va = lds_afrag(vr);
      acc[0][dt] = __builtin_amdgcn_wmma_f32_16x16x32_f16(false, va, false, pb[0], (short)0, acc[0][dt], false, false);
      acc[1][dt] = __builtin_amdgcn_wmma_f32_16x16x32_f16(false, va, false, pb[1], (short)0, acc[1][dt], false, false);
    }

    // ---- convert + store NEXT chunk into the other buffer, then flip ----
    if (has_next) stage_lds_store(&sK[buf ^ 1][0], &sV[buf ^ 1][0], tid, nk, nv);
    __syncthreads();
  }

  // ---- epilogue: normalize and store (d = dt*16 + 8*sub + j, q = tile base + l16) ----
#pragma unroll
  for (int t = 0; t < QTILES; ++t) {
    const float inv_l = 1.0f / l_i[t];
    float* orow = Og + ((size_t)bh * SEQ + qbase + 16 * t + l16) * DH + 8 * sub;
#pragma unroll
    for (int dt = 0; dt < 4; ++dt)
#pragma unroll
      for (int j = 0; j < 8; ++j)
        orow[dt * 16 + j] = acc[t][dt][j] * inv_l;
  }
}

extern "C" void kernel_launch(void* const* d_in, const int* in_sizes, int n_in,
                              void* d_out, int out_size, void* d_ws, size_t ws_size,
                              hipStream_t stream) {
  const float* q = (const float*)d_in[0];
  const float* k = (const float*)d_in[1];
  const float* v = (const float*)d_in[2];
  const float* m = (const float*)d_in[3];
  float* o = (float*)d_out;
  (void)in_sizes; (void)n_in; (void)out_size; (void)d_ws; (void)ws_size;

  dim3 grid(BH * (SEQ / QB));
  dim3 block(NWAVES * 32);
  hipLaunchKernelGGL(fa_fwd_wmma_25151328485506, grid, block, 0, stream, q, k, v, m, o);
}
